// Attention_74775380623443
// MI455X (gfx1250) — matrix operand order repaired and checked by reading, not compiled, not run on hardware
//
#include <hip/hip_runtime.h>
#include <float.h>
#include <math.h>

// ---------------------------------------------------------------------------
// Cosine-sim causal flash attention for MI455X (gfx1250, wave32, WMMA).
// b=2, h=16, n=2048, d=64. fp32 I/O, f16 WMMA matmuls, f32 softmax.
// Both batches share one workgroup so the 268MB attn_bias stream (the HBM
// roofline bottleneck) is read from DRAM once, not twice.
// ---------------------------------------------------------------------------

typedef __attribute__((ext_vector_type(16))) _Float16 v16h;
typedef __attribute__((ext_vector_type(8)))  float    v8f;

#define NCTX 2048
#define NH   16
#define HD   64      // head dim
#define TMB  64      // query rows per batch per workgroup (4 waves x 16 rows)
#define TJ   64      // kv columns per iteration
#define LDSS 72      // LDS row stride in f16 elements (pad vs bank conflicts)
#define NEGF (-3.0e38f)

union F16x16 { v16h v; _Float16 h[16]; unsigned int u[8]; };

// A-operand fragment (16x32 f16): K = (lane>=16?16:0) + 2*v + h
__device__ __forceinline__ v16h load_afrag(const _Float16* rowbase, int lane, int kch) {
  F16x16 t;
  const int kb = ((lane >> 4) << 4) + (kch << 5);
#pragma unroll
  for (int vv = 0; vv < 8; ++vv)
    t.u[vv] = *(const unsigned int*)(rowbase + kb + (vv << 1));
  return t.v;
}

// B-operand fragment (32x16 f16): N = lane%16 (selected by rowbase),
// K = (lane>=16?16:0) + 2*v + h  (contiguous pairs along the stored row)
__device__ __forceinline__ v16h load_bfrag(const _Float16* rowbase, int lane, int kch) {
  F16x16 t;
  const int kb = ((lane >> 4) << 4) + (kch << 5);
#pragma unroll
  for (int vv = 0; vv < 8; ++vv)
    t.u[vv] = *(const unsigned int*)(rowbase + kb + (vv << 1));
  return t.v;
}

__global__ __launch_bounds__(256, 1)
void cosine_attn_kernel(const float* __restrict__ Q, const float* __restrict__ K,
                        const float* __restrict__ V, const unsigned char* __restrict__ MASK,
                        const float* __restrict__ BIAS, const float* __restrict__ QS,
                        const float* __restrict__ KS, float* __restrict__ OUT)
{
  // rows 0..63 = batch 0, rows 64..127 = batch 1 in each tile buffer
  __shared__ _Float16 sQ [2 * TMB * LDSS];  // q, normalized+scaled, f16 [row][d]
  __shared__ _Float16 sKP[2 * TJ  * LDSS];  // k tile [j][d]; reused as P scratch after QK
  __shared__ _Float16 sVt[2 * HD  * LDSS];  // v tile transposed [d][j]

  const int i0   = blockIdx.x * TMB;
  const int h    = blockIdx.y;
  const int tid  = threadIdx.x;
  const int lane = tid & 31;
  const int wave = tid >> 5;            // 0..7
  const int wb   = wave >> 2;           // batch of this wave (0/1)
  const int wq   = wave & 3;            // 16-row block within batch tile
  const int l15  = lane & 15;
  const int grp8 = (lane >> 4) << 3;    // 0 or 8

  // ---- preprocess Q tiles (both batches): l2norm * q_scale -> f16 LDS ----
  {
    const int row = tid >> 1;           // 0..127 (batch*64 + local row)
    const int seg = tid & 1;            // 32 elements each
    const int qb  = row >> 6;
    const int qr  = row & 63;
    const float* qp = Q + (((size_t)(qb * NH + h)) * NCTX + (i0 + qr)) * HD + seg * 32;
    float x[32];
    float ss = 0.f;
#pragma unroll
    for (int e = 0; e < 32; e += 4) {
      float4 t = *(const float4*)(qp + e);
      x[e] = t.x; x[e + 1] = t.y; x[e + 2] = t.z; x[e + 3] = t.w;
      ss += t.x * t.x + t.y * t.y + t.z * t.z + t.w * t.w;
    }
    ss += __shfl_xor(ss, 1, 32);
    const float inv = 1.0f / fmaxf(sqrtf(ss), 1e-12f);
#pragma unroll
    for (int e = 0; e < 32; ++e)
      sQ[row * LDSS + seg * 32 + e] = (_Float16)(x[e] * inv * QS[seg * 32 + e]);
  }
  __syncthreads();

  // Loop-invariant Q A-fragments (wave*16 == wb*64 + wq*16)
  const _Float16* qrow = &sQ[(wave * 16 + l15) * LDSS];
  const v16h qa0 = load_afrag(qrow, lane, 0);
  const v16h qa1 = load_afrag(qrow, lane, 1);

  float m[8], l[8];
  v8f o[4];
#pragma unroll
  for (int r = 0; r < 8; ++r) { m[r] = NEGF; l[r] = 0.f; }
#pragma unroll
  for (int n = 0; n < 4; ++n) o[n] = (v8f){};

  const unsigned char* maskp = MASK + (size_t)wb * NCTX;
  const float* biasrow = BIAS + ((size_t)h * NCTX + (i0 + wq * 16 + grp8)) * NCTX;
  const int jend = i0 + TMB;            // causal: only j <= i needed

  for (int j0 = 0; j0 < jend; j0 += TJ) {
    __syncthreads();                    // protect sKP/sVt reuse
    {
      const int row = tid >> 1;         // 0..127 (batch*64 + kv row)
      const int seg = tid & 1;          // 32 elements each
      const int kb  = row >> 6;
      const int kr  = row & 63;
      const size_t base = (((size_t)(kb * NH + h)) * NCTX + (j0 + kr)) * HD + seg * 32;
      const float* kp = K + base;
      const float* vp = V + base;
      float x[32];
      float ss = 0.f;
#pragma unroll
      for (int e = 0; e < 32; e += 4) {
        float4 t = *(const float4*)(kp + e);
        x[e] = t.x; x[e + 1] = t.y; x[e + 2] = t.z; x[e + 3] = t.w;
        ss += t.x * t.x + t.y * t.y + t.z * t.z + t.w * t.w;
      }
      ss += __shfl_xor(ss, 1, 32);
      const float inv = 1.0f / fmaxf(sqrtf(ss), 1e-12f);
#pragma unroll
      for (int e = 0; e < 32; ++e)
        sKP[row * LDSS + seg * 32 + e] = (_Float16)(x[e] * inv * KS[seg * 32 + e]);
#pragma unroll
      for (int e = 0; e < 32; e += 4) {
        float4 t = *(const float4*)(vp + e);
        sVt[(kb * HD + seg * 32 + e + 0) * LDSS + kr] = (_Float16)t.x;
        sVt[(kb * HD + seg * 32 + e + 1) * LDSS + kr] = (_Float16)t.y;
        sVt[(kb * HD + seg * 32 + e + 2) * LDSS + kr] = (_Float16)t.z;
        sVt[(kb * HD + seg * 32 + e + 3) * LDSS + kr] = (_Float16)t.w;
      }
    }
    __syncthreads();

    // prefetch next j-tile's bias rows (the dominant HBM stream)
    if (l15 == 0 && (j0 + TJ) < jend) {
      const float* bnext = biasrow + (j0 + TJ);
#pragma unroll
      for (int r = 0; r < 8; ++r) {
        __builtin_prefetch(bnext + (size_t)r * NCTX,      0, 3);
        __builtin_prefetch(bnext + (size_t)r * NCTX + 32, 0, 3);
      }
    }

    // ---- S = Q * K^T (f32 accum), 4 N-subtiles x 2 K-chunks = 8 WMMA ----
    v8f s[4];
#pragma unroll
    for (int n16 = 0; n16 < 4; ++n16) {
      const _Float16* krow = &sKP[(wb * TJ + n16 * 16 + l15) * LDSS];
      v16h b0 = load_bfrag(krow, lane, 0);
      v16h b1 = load_bfrag(krow, lane, 1);
      v8f c = (v8f){};
      c = __builtin_amdgcn_wmma_f32_16x16x32_f16(false, qa0, false, b0, (short)0, c, false, false);
      c = __builtin_amdgcn_wmma_f32_16x16x32_f16(false, qa1, false, b1, (short)0, c, false, false);
      s[n16] = c;
    }
    __syncthreads();                    // all waves done reading sKP -> reuse as P

    // ---- scale*8 + bias, key mask, causal mask (f32) ----
#pragma unroll
    for (int n16 = 0; n16 < 4; ++n16) {
      const int jg = j0 + n16 * 16 + l15;
      const bool mk = maskp[jg] != 0;
      const float* brow = biasrow + jg;
#pragma unroll
      for (int r = 0; r < 8; ++r) {
        const int ig = i0 + wq * 16 + grp8 + r;
        float x = s[n16][r] * 8.0f + brow[(size_t)r * NCTX];
        if (!mk)     x = NEGF;
        if (jg > ig) x = NEGF;
        s[n16][r] = x;
      }
    }

    // ---- online softmax: row max / exp / row sum / rescale ----
    float mn[8], fac[8], ls[8];
#pragma unroll
    for (int r = 0; r < 8; ++r) {
      float vmax = fmaxf(fmaxf(s[0][r], s[1][r]), fmaxf(s[2][r], s[3][r]));
      vmax = fmaxf(vmax, __shfl_xor(vmax, 1, 32));
      vmax = fmaxf(vmax, __shfl_xor(vmax, 2, 32));
      vmax = fmaxf(vmax, __shfl_xor(vmax, 4, 32));
      vmax = fmaxf(vmax, __shfl_xor(vmax, 8, 32));
      mn[r]  = fmaxf(m[r], vmax);
      fac[r] = __expf(m[r] - mn[r]);
      m[r]   = mn[r];
      ls[r]  = 0.f;
    }
    _Float16* pbase = &sKP[wave * 16 * LDSS];   // P scratch aliases this wave's K rows
#pragma unroll
    for (int n16 = 0; n16 < 4; ++n16) {
#pragma unroll
      for (int r = 0; r < 8; ++r) {
        const float e = __expf(s[n16][r] - mn[r]);
        ls[r] += e;
        pbase[(grp8 + r) * LDSS + n16 * 16 + l15] = (_Float16)e;
      }
    }
#pragma unroll
    for (int r = 0; r < 8; ++r) {
      float t = ls[r];
      t += __shfl_xor(t, 1, 32);
      t += __shfl_xor(t, 2, 32);
      t += __shfl_xor(t, 4, 32);
      t += __shfl_xor(t, 8, 32);
      l[r] = l[r] * fac[r] + t;
    }
#pragma unroll
    for (int n = 0; n < 4; ++n)
#pragma unroll
      for (int r = 0; r < 8; ++r)
        o[n][r] *= fac[r];

    // order same-wave LDS write->read of the P scratch
    asm volatile("s_wait_dscnt 0" ::: "memory");

    // ---- O += P * V (8 WMMA) ----
    const _Float16* prow = &sKP[(wave * 16 + l15) * LDSS];
    const v16h pa0 = load_afrag(prow, lane, 0);
    const v16h pa1 = load_afrag(prow, lane, 1);
#pragma unroll
    for (int nd = 0; nd < 4; ++nd) {
      const _Float16* vrow = &sVt[(wb * HD + nd * 16 + l15) * LDSS];
      v16h b0 = load_bfrag(vrow, lane, 0);
      v16h b1 = load_bfrag(vrow, lane, 1);
      o[nd] = __builtin_amdgcn_wmma_f32_16x16x32_f16(false, pa0, false, b0, (short)0, o[nd], false, false);
      o[nd] = __builtin_amdgcn_wmma_f32_16x16x32_f16(false, pa1, false, b1, (short)0, o[nd], false, false);
    }
  }

  // ---- epilogue: O / l -> global ----
#pragma unroll
  for (int nd = 0; nd < 4; ++nd) {
#pragma unroll
    for (int r = 0; r < 8; ++r) {
      const int ig = i0 + wq * 16 + grp8 + r;
      const int dc = nd * 16 + l15;
      OUT[(((size_t)(wb * NH + h)) * NCTX + ig) * HD + dc] = o[nd][r] / l[r];
    }
  }
}

extern "C" void kernel_launch(void* const* d_in, const int* in_sizes, int n_in,
                              void* d_out, int out_size, void* d_ws, size_t ws_size,
                              hipStream_t stream) {
  (void)in_sizes; (void)n_in; (void)out_size; (void)d_ws; (void)ws_size;
  const float*         q    = (const float*)d_in[0];
  const float*         k    = (const float*)d_in[1];
  const float*         v    = (const float*)d_in[2];
  const unsigned char* mask = (const unsigned char*)d_in[3];
  const float*         bias = (const float*)d_in[4];
  const float*         qs   = (const float*)d_in[5];
  const float*         ks   = (const float*)d_in[6];
  float*               out  = (float*)d_out;

  dim3 grid(NCTX / TMB, NH, 1);
  dim3 block(256);
  cosine_attn_kernel<<<grid, block, 0, stream>>>(q, k, v, mask, bias, qs, ks, out);
}
